// TriangleAttentionMQ_68178310857260
// MI455X (gfx1250) — matrix operand order repaired and checked by reading, not compiled, not run on hardware
//
#include <hip/hip_runtime.h>

typedef __attribute__((ext_vector_type(16))) __bf16 v16bf;
typedef __attribute__((ext_vector_type(8)))  __bf16 v8bf;
typedef __attribute__((ext_vector_type(8)))  float  v8f;
typedef int v4i_ __attribute__((vector_size(16)));

#define N_    160
#define E_    256
#define D_    64
#define H_    4
#define NN_   (N_*N_)
#define NCOLS 784
#define SCALE_F 0.125f

#if defined(__has_builtin)
#if __has_builtin(__builtin_amdgcn_global_load_async_to_lds_b128) && \
    __has_builtin(__builtin_amdgcn_s_wait_asynccnt)
#define HAVE_ASYNC_LDS 1
#endif
#endif
#ifndef HAVE_ASYNC_LDS
#define HAVE_ASYNC_LDS 0
#endif

__device__ inline v8f wmma_bf16(v16bf a, v16bf b, v8f c) {
  return __builtin_amdgcn_wmma_f32_16x16x32_bf16(false, a, false, b, (short)0, c, false, false);
}
__device__ inline v16bf mk16(v8bf lo, v8bf hi) {
  return __builtin_shufflevector(lo, hi, 0,1,2,3,4,5,6,7,8,9,10,11,12,13,14,15);
}

// 16-byte global -> LDS copy; async-to-LDS path (ASYNCcnt-tracked) on gfx1250.
__device__ inline void stage16(const v8bf* g, v8bf* l) {
#if HAVE_ASYNC_LDS
  __builtin_amdgcn_global_load_async_to_lds_b128(
      (__attribute__((address_space(1))) v4i_*)(void*)g,
      (__attribute__((address_space(3))) v4i_*)l, 0, 0);
#else
  *l = *g;
#endif
}
__device__ inline void stage_wait() {
#if HAVE_ASYNC_LDS
  __builtin_amdgcn_s_wait_asynccnt(0);
#endif
}

// ---------------- weight packing ----------------
__global__ void prep_weights(const float* __restrict__ Wq_in, const float* __restrict__ bq_in,
                             const float* __restrict__ Wkv_in, const float* __restrict__ bkv_in,
                             const float* __restrict__ Weg_in, const float* __restrict__ beg_in,
                             const float* __restrict__ Wq_out, const float* __restrict__ bq_out,
                             const float* __restrict__ Wkv_out, const float* __restrict__ bkv_out,
                             const float* __restrict__ Weg_out, const float* __restrict__ beg_out,
                             const float* __restrict__ Wo,
                             __bf16* __restrict__ Wcat, float* __restrict__ bcat,
                             __bf16* __restrict__ WoT) {
  int id = blockIdx.x * blockDim.x + threadIdx.x;
  const int NW = NCOLS * E_;
  if (id < NW) {
    int col = id / E_, k = id % E_;
    float v, bv;
    if (col < 256)      { int h = col/64, d = col%64; v = Wq_in [k*256 + d*4 + h]*SCALE_F; bv = bq_in [d*4+h]*SCALE_F; }
    else if (col < 512) { int c = col-256; int h=c/64, d=c%64; v = Wq_out[k*256 + d*4 + h]*SCALE_F; bv = bq_out[d*4+h]*SCALE_F; }
    else if (col < 576) { int d = col-512; v = Wkv_in [k*128 + d];      bv = bkv_in [d];    }
    else if (col < 640) { int d = col-576; v = Wkv_in [k*128 + 64 + d]; bv = bkv_in [64+d]; }
    else if (col < 704) { int d = col-640; v = Wkv_out[k*128 + d];      bv = bkv_out[d];    }
    else if (col < 768) { int d = col-704; v = Wkv_out[k*128 + 64 + d]; bv = bkv_out[64+d]; }
    else if (col < 776) { int c = col-768; v = Weg_in [k*8 + c];        bv = beg_in [c];    }
    else                { int c = col-776; v = Weg_out[k*8 + c];        bv = beg_out[c];    }
    Wcat[col*E_ + k] = (__bf16)v;
    if (k == 0) bcat[col] = bv;
  } else {
    int id2 = id - NW;                       // WoT[o][k'] with k' = hh*64+d <-> Wo row d*8+hh
    if (id2 < 256*512) {
      int o = id2 / 512, kp = id2 % 512;
      int d = kp % 64, hh = kp / 64;
      WoT[o*512 + kp] = (__bf16)Wo[(d*8 + hh)*256 + o];
    }
  }
}

// ---------------- layernorm ----------------
__global__ void ln_kernel(const float* __restrict__ e, const float* __restrict__ g,
                          const float* __restrict__ b, __bf16* __restrict__ out) {
  __shared__ float red[256];
  int row = blockIdx.x, t = threadIdx.x;
  float x = e[(size_t)row*E_ + t];
  red[t] = x; __syncthreads();
  for (int s = 128; s > 0; s >>= 1) { if (t < s) red[t] += red[t+s]; __syncthreads(); }
  float mu = red[0] * (1.0f/256.0f);
  __syncthreads();
  float dx = x - mu;
  red[t] = dx*dx; __syncthreads();
  for (int s = 128; s > 0; s >>= 1) { if (t < s) red[t] += red[t+s]; __syncthreads(); }
  float var = red[0] * (1.0f/256.0f);
  float r = rsqrtf(var + 1e-5f);
  out[(size_t)row*E_ + t] = (__bf16)(dx * r * g[t] + b[t]);
}

// ---------------- generic bf16 WMMA GEMM: C[row][col] = A[row][K] * WT[col][K] + bias ----------------
__global__ __launch_bounds__(256) void gemm_bf16(const __bf16* __restrict__ A,
                                                 const __bf16* __restrict__ WT,
                                                 const float* __restrict__ bias,
                                                 float* __restrict__ C,
                                                 int ncols, int K) {
  extern __shared__ __bf16 At[];             // 16 rows * K
  int row0 = blockIdx.x * 16;
  int t = threadIdx.x;
  int cnt = 16*K/8;
  const v8bf* src = (const v8bf*)(A + (size_t)row0 * K);
  v8bf* dst = (v8bf*)At;
  for (int i = t; i < cnt; i += 256) stage16(src + i, dst + i);
  stage_wait();
  __syncthreads();
  int w = t >> 5, lane = t & 31;
  int ct = blockIdx.y * 8 + w;
  if (ct * 16 >= ncols) return;
  int ln16 = lane & 15, hl = lane >> 4;
  v8f acc = {};
  const __bf16* arow = At + ln16 * K;
  const __bf16* brow = WT + (size_t)(ct*16 + ln16) * K;
  __builtin_prefetch(brow, 0, 3);
  for (int kb = 0; kb < K; kb += 32) {
    v8bf lo = *(const v8bf*)(arow + kb + hl*16);
    v8bf hi = *(const v8bf*)(arow + kb + hl*16 + 8);
    v16bf a = mk16(lo, hi);
    v16bf b = *(const v16bf*)(brow + kb + hl*16);
    acc = wmma_bf16(a, b, acc);
  }
  int col = ct*16 + ln16;
  float bv = bias ? bias[col] : 0.0f;
#pragma unroll
  for (int r = 0; r < 8; ++r) {
    int row = row0 + r + hl*8;
    C[(size_t)row * ncols + col] = acc[r] + bv;
  }
}

// ---------------- rearranges ----------------
__global__ void rearrange_q(const float* __restrict__ Cp, __bf16* __restrict__ Qbf) {
  int id = blockIdx.x*256 + threadIdx.x;     // (((s*4+h)*160 + j)*160 + i)*64 + d
  int d  = id & 63;
  int i  = (id >> 6) % N_;
  int jj = (id / (64*N_)) % N_;
  int hs = id / (64*N_*N_);
  int h = hs & 3, s = hs >> 2;
  Qbf[id] = (__bf16)Cp[(size_t)(i*N_ + jj)*NCOLS + s*256 + h*64 + d];
}

__global__ void rearrange_kv(const float* __restrict__ Cp,
                             __bf16* __restrict__ Kin,  __bf16* __restrict__ VTin,
                             __bf16* __restrict__ Kout, __bf16* __restrict__ VTout) {
  int id = blockIdx.x*256 + threadIdx.x;     // 4 * 160*160*64
  int a   = id / (N_*N_*D_);
  int rem = id % (N_*N_*D_);
  int j   = rem / (N_*D_);
  int r2  = rem % (N_*D_);
  if (a == 0 || a == 2) {                    // K layouts [j][k][d]
    int k = r2 / D_, d = r2 % D_;
    int srow = (a == 0) ? (j*N_ + k) : (k*N_ + j);
    int scol = (a == 0) ? (512 + d) : (640 + d);
    ((a == 0) ? Kin : Kout)[(j*N_ + k)*D_ + d] = (__bf16)Cp[(size_t)srow*NCOLS + scol];
  } else {                                   // V transposed [j][d][k]
    int d = r2 / N_, k = r2 % N_;
    int srow = (a == 1) ? (j*N_ + k) : (k*N_ + j);
    int scol = (a == 1) ? (576 + d) : (704 + d);
    ((a == 1) ? VTin : VTout)[(j*D_ + d)*N_ + k] = (__bf16)Cp[(size_t)srow*NCOLS + scol];
  }
}

__global__ void rearrange_bg(const float* __restrict__ Cp, const float* __restrict__ mask,
                             float* __restrict__ biasI, float* __restrict__ gateI,
                             float* __restrict__ biasO, float* __restrict__ gateO) {
  int id = blockIdx.x*256 + threadIdx.x;     // 4 * H * N * N
  int a   = id / (H_*N_*N_);
  int rem = id % (H_*N_*N_);
  int h   = rem / (N_*N_);
  int ik  = rem % (N_*N_);
  int i = ik / N_, k = ik % N_;
  if (a == 0)      { biasI[rem] = Cp[(size_t)(i*N_+k)*NCOLS + 768 + h] + mask[i*N_+k]; }
  else if (a == 1) { float x = Cp[(size_t)(i*N_+k)*NCOLS + 772 + h] + mask[i*N_+k];
                     gateI[rem] = 1.0f/(1.0f + __expf(-x)); }
  else if (a == 2) { biasO[rem] = Cp[(size_t)(k*N_+i)*NCOLS + 776 + h] + mask[k*N_+i]; }
  else             { float x = Cp[(size_t)(k*N_+i)*NCOLS + 780 + h] + mask[k*N_+i];
                     gateO[rem] = 1.0f/(1.0f + __expf(-x)); }
}

// ---------------- attention: one (j, h, branch) per block, 10 waves ----------------
__global__ __launch_bounds__(320) void attn_kernel(
    const __bf16* __restrict__ Qbf,
    const __bf16* __restrict__ Kin,  const __bf16* __restrict__ VTin,
    const __bf16* __restrict__ Kout, const __bf16* __restrict__ VTout,
    const float* __restrict__ biasI, const float* __restrict__ gateI,
    const float* __restrict__ biasO, const float* __restrict__ gateO,
    __bf16* __restrict__ Va) {
  extern __shared__ __bf16 smem[];
  __bf16* Klds  = smem;                 // 160*64
  __bf16* Vtlds = smem + N_*D_;         // 64*160
  __bf16* probs = smem + 2*N_*D_;       // 160*160
  int j = blockIdx.x, h = blockIdx.y, s = blockIdx.z;
  int t = threadIdx.x, w = t >> 5, lane = t & 31;
  int ln16 = lane & 15, hl = lane >> 4;

  const __bf16* Kg = ((s == 0) ? Kin  : Kout ) + (size_t)j*N_*D_;
  const __bf16* Vg = ((s == 0) ? VTin : VTout) + (size_t)j*D_*N_;
  const float* bias = ((s == 0) ? biasI : biasO) + (size_t)h*N_*N_;
  const float* gate = ((s == 0) ? gateI : gateO) + (size_t)h*N_*N_;
  const __bf16* Qbase = Qbf + (size_t)((s*4 + h)*N_ + j)*N_*D_;

  { // stage K, V^T into LDS (async-to-LDS path on gfx1250)
    const v8bf* ks = (const v8bf*)Kg; v8bf* kd = (v8bf*)Klds;
    const v8bf* vs = (const v8bf*)Vg; v8bf* vd = (v8bf*)Vtlds;
    for (int i2 = t; i2 < N_*D_/8; i2 += 320) { stage16(ks + i2, kd + i2); stage16(vs + i2, vd + i2); }
  }
  stage_wait();
  __syncthreads();

  // Q A-fragments for this wave's 16 rows
  const __bf16* qrow = Qbase + (size_t)(w*16 + ln16)*D_;
  v16bf aq[2];
#pragma unroll
  for (int ks2 = 0; ks2 < 2; ++ks2) {
    v8bf lo = *(const v8bf*)(qrow + ks2*32 + hl*16);
    v8bf hi = *(const v8bf*)(qrow + ks2*32 + hl*16 + 8);
    aq[ks2] = mk16(lo, hi);
  }

  // scores S = Q K^T + bias(+mask)
  v8f sc[10];
#pragma unroll
  for (int nb = 0; nb < 10; ++nb) {
    v8f acc = {};
#pragma unroll
    for (int ks2 = 0; ks2 < 2; ++ks2) {
      v16bf b = *(const v16bf*)(Klds + (nb*16 + ln16)*D_ + ks2*32 + hl*16);
      acc = wmma_bf16(aq[ks2], b, acc);
    }
    int colg = nb*16 + ln16;
#pragma unroll
    for (int r = 0; r < 8; ++r) {
      int rowg = w*16 + r + hl*8;
      acc[r] += bias[rowg*N_ + colg];
    }
    sc[nb] = acc;
  }

  // row softmax (rows live in 16-lane half-groups; xor masks 1,2,4,8 stay in-group)
  float inv[8];
#pragma unroll
  for (int r = 0; r < 8; ++r) {
    float m = sc[0][r];
#pragma unroll
    for (int nb = 1; nb < 10; ++nb) m = fmaxf(m, sc[nb][r]);
    m = fmaxf(m, __shfl_xor(m, 1, 32));
    m = fmaxf(m, __shfl_xor(m, 2, 32));
    m = fmaxf(m, __shfl_xor(m, 4, 32));
    m = fmaxf(m, __shfl_xor(m, 8, 32));
    float sum = 0.0f;
#pragma unroll
    for (int nb = 0; nb < 10; ++nb) { float e2 = __expf(sc[nb][r] - m); sc[nb][r] = e2; sum += e2; }
    sum += __shfl_xor(sum, 1, 32);
    sum += __shfl_xor(sum, 2, 32);
    sum += __shfl_xor(sum, 4, 32);
    sum += __shfl_xor(sum, 8, 32);
    inv[r] = 1.0f / sum;
  }

  // gate, convert to bf16, stage probs in LDS (wave's own 16 rows)
#pragma unroll
  for (int nb = 0; nb < 10; ++nb) {
    int colg = nb*16 + ln16;
#pragma unroll
    for (int r = 0; r < 8; ++r) {
      int rowg = w*16 + r + hl*8;
      float pv = sc[nb][r] * inv[r] * gate[rowg*N_ + colg];
      probs[rowg*N_ + colg] = (__bf16)pv;
    }
  }
  __syncthreads();

  // P @ V  (160x64 out per block, 16x64 per wave)
  v8f ov[4] = { {}, {}, {}, {} };
  const __bf16* prow = probs + (w*16 + ln16)*N_;
#pragma unroll
  for (int ks2 = 0; ks2 < 5; ++ks2) {
    v8bf lo = *(const v8bf*)(prow + ks2*32 + hl*16);
    v8bf hi = *(const v8bf*)(prow + ks2*32 + hl*16 + 8);
    v16bf a = mk16(lo, hi);
#pragma unroll
    for (int db = 0; db < 4; ++db) {
      v16bf b = *(const v16bf*)(Vtlds + (db*16 + ln16)*N_ + ks2*32 + hl*16);
      ov[db] = wmma_bf16(a, b, ov[db]);
    }
  }

  int hh = s*4 + h;
#pragma unroll
  for (int db = 0; db < 4; ++db) {
    int d = db*16 + ln16;
#pragma unroll
    for (int r = 0; r < 8; ++r) {
      int ig = w*16 + r + hl*8;
      Va[(size_t)(ig*N_ + j)*512 + hh*64 + d] = (__bf16)ov[db][r];
    }
  }
}

// ---------------- host ----------------
extern "C" void kernel_launch(void* const* d_in, const int* in_sizes, int n_in,
                              void* d_out, int out_size, void* d_ws, size_t ws_size,
                              hipStream_t stream) {
  (void)in_sizes; (void)n_in; (void)out_size; (void)ws_size;
  const float* e      = (const float*)d_in[0];
  const float* mask   = (const float*)d_in[1];
  const float* ln_g   = (const float*)d_in[2];
  const float* ln_b   = (const float*)d_in[3];
  const float* Wq_in  = (const float*)d_in[4];
  const float* bq_in  = (const float*)d_in[5];
  const float* Wkv_in = (const float*)d_in[6];
  const float* bkv_in = (const float*)d_in[7];
  const float* Weg_in = (const float*)d_in[8];
  const float* beg_in = (const float*)d_in[9];
  const float* Wq_out  = (const float*)d_in[10];
  const float* bq_out  = (const float*)d_in[11];
  const float* Wkv_out = (const float*)d_in[12];
  const float* bkv_out = (const float*)d_in[13];
  const float* Weg_out = (const float*)d_in[14];
  const float* beg_out = (const float*)d_in[15];
  const float* Wo = (const float*)d_in[16];
  const float* bo = (const float*)d_in[17];
  float* out = (float*)d_out;

  char* p = (char*)d_ws;
  auto alloc = [&](size_t bytes) -> char* {
    char* r = p; p += (bytes + 255) & ~(size_t)255; return r;
  };
  __bf16* e_ln  = (__bf16*)alloc((size_t)NN_*E_*2);
  __bf16* Wcat  = (__bf16*)alloc((size_t)NCOLS*E_*2);
  float*  bcat  = (float*) alloc((size_t)NCOLS*4);
  __bf16* WoT   = (__bf16*)alloc((size_t)256*512*2);
  float*  Cproj = (float*) alloc((size_t)NN_*NCOLS*4);
  __bf16* Qbf   = (__bf16*)alloc((size_t)2*H_*N_*N_*D_*2);
  __bf16* Kin   = (__bf16*)alloc((size_t)N_*N_*D_*2);
  __bf16* VTin  = (__bf16*)alloc((size_t)N_*N_*D_*2);
  __bf16* Kout  = (__bf16*)alloc((size_t)N_*N_*D_*2);
  __bf16* VTout = (__bf16*)alloc((size_t)N_*N_*D_*2);
  float*  biasI = (float*) alloc((size_t)H_*N_*N_*4);
  float*  gateI = (float*) alloc((size_t)H_*N_*N_*4);
  float*  biasO = (float*) alloc((size_t)H_*N_*N_*4);
  float*  gateO = (float*) alloc((size_t)H_*N_*N_*4);
  __bf16* Vabf  = (__bf16*)alloc((size_t)NN_*512*2);

  int prepThreads = NCOLS*E_ + 256*512;
  prep_weights<<<(prepThreads + 255)/256, 256, 0, stream>>>(
      Wq_in, bq_in, Wkv_in, bkv_in, Weg_in, beg_in,
      Wq_out, bq_out, Wkv_out, bkv_out, Weg_out, beg_out, Wo,
      Wcat, bcat, WoT);

  ln_kernel<<<NN_, 256, 0, stream>>>(e, ln_g, ln_b, e_ln);

  gemm_bf16<<<dim3(NN_/16, 7), 256, 16*E_*2, stream>>>(e_ln, Wcat, bcat, Cproj, NCOLS, E_);

  rearrange_q <<<(2*H_*N_*N_*D_)/256, 256, 0, stream>>>(Cproj, Qbf);
  rearrange_kv<<<(4*N_*N_*D_)/256, 256, 0, stream>>>(Cproj, Kin, VTin, Kout, VTout);
  rearrange_bg<<<(4*H_*N_*N_)/256, 256, 0, stream>>>(Cproj, mask, biasI, gateI, biasO, gateO);

  attn_kernel<<<dim3(N_, H_, 2), 320, (2*N_*D_ + N_*N_)*2, stream>>>(
      Qbf, Kin, VTin, Kout, VTout, biasI, gateI, biasO, gateO, Vabf);

  gemm_bf16<<<dim3(NN_/16, 2), 256, 16*512*2, stream>>>(Vabf, WoT, bo, out, 256, 512);
}
